// DeformableDepthwiseSeparableConv_13237089206821
// MI455X (gfx1250) — matrix operand order repaired and checked by reading, not compiled, not run on hardware
//
#include <hip/hip_runtime.h>
#include <hip/hip_bf16.h>

// Problem constants (match reference)
static constexpr int B   = 4;
static constexpr int T   = 4096;
static constexpr int C   = 512;
static constexpr int OUT = 512;
static constexpr int K   = 11;
static constexpr int PL  = 5;           // left reflect pad
static constexpr int TP  = T + 10;      // padded length

typedef __attribute__((ext_vector_type(16))) _Float16 v16h;
typedef __attribute__((ext_vector_type(8)))  _Float16 v8h;
typedef __attribute__((ext_vector_type(8)))  float    v8f;

__device__ __forceinline__ int reflect_idx(int m) {
    m = (m < 0) ? -m : m;
    m = (m >= T) ? (2 * T - 2 - m) : m;
    return m;
}

// ---------------------------------------------------------------------------
// Kernel 0: convert w_pw (OUT x C, f32) -> f16, and zero the stats accumulators
// ---------------------------------------------------------------------------
__global__ __launch_bounds__(256) void k_prep(const float* __restrict__ wpw,
                                              _Float16* __restrict__ wh,
                                              float* __restrict__ stats) {
    int i = blockIdx.x * 256 + threadIdx.x;
    if (i < OUT * C) wh[i] = (_Float16)wpw[i];
    if (blockIdx.x == 0 && threadIdx.x < 8) stats[threadIdx.x] = 0.0f;
}

// ---------------------------------------------------------------------------
// Kernel 1: offsets.  off[b,t,k] = sum_c w_off_p[k,c] * (sum_j w_off_d[c,j] *
//           x[b, reflect(t+j-5), c]).  One wave per timestep, lanes stride C.
// ---------------------------------------------------------------------------
__global__ __launch_bounds__(256) void k_offsets(const float* __restrict__ x,
                                                 const float* __restrict__ wod,
                                                 const float* __restrict__ wop,
                                                 float* __restrict__ off) {
    const int wave = threadIdx.x >> 5;
    const int lane = threadIdx.x & 31;
    const int b  = blockIdx.x / (T / 64);
    const int t0 = (blockIdx.x % (T / 64)) * 64;

    for (int it = 0; it < 8; ++it) {
        const int t = t0 + it * 8 + wave;
        float acc[K];
#pragma unroll
        for (int k = 0; k < K; ++k) acc[k] = 0.0f;

        for (int c = lane; c < C; c += 32) {
            float v = 0.0f;
#pragma unroll
            for (int j = 0; j < K; ++j) {
                const int m = reflect_idx(t + j - PL);
                v = fmaf(wod[c * K + j], x[((size_t)b * T + m) * C + c], v);
            }
#pragma unroll
            for (int k = 0; k < K; ++k)
                acc[k] = fmaf(wop[k * C + c], v, acc[k]);
        }
        // wave32 butterfly reduction
#pragma unroll
        for (int k = 0; k < K; ++k) {
#pragma unroll
            for (int m = 16; m >= 1; m >>= 1)
                acc[k] += __shfl_xor(acc[k], m, 32);
        }
        if (lane == 0) {
#pragma unroll
            for (int k = 0; k < K; ++k)
                off[((size_t)b * T + t) * K + k] = acc[k];
        }
    }
}

// ---------------------------------------------------------------------------
// Kernel 2: deformable sample + depthwise combine + PReLU, y (f32) to scratch,
//           accumulate per-batch sum/sumsq via atomics.
// ---------------------------------------------------------------------------
__global__ __launch_bounds__(256) void k_sample(const float* __restrict__ x,
                                                const float* __restrict__ off,
                                                const float* __restrict__ wdw,
                                                const float* __restrict__ bdw,
                                                const float* __restrict__ pa,
                                                float* __restrict__ y,
                                                float* __restrict__ stats) {
    const int TT = 16;
    const int b  = blockIdx.x / (T / TT);
    const int t0 = (blockIdx.x % (T / TT)) * TT;
    const int tid = threadIdx.x;

    __shared__ int   sm0[TT][K];
    __shared__ int   sm1[TT][K];
    __shared__ float sfr[TT][K];
    __shared__ float swdw[C][K];
    __shared__ float sbdw[C];
    __shared__ float sred[256];

    for (int i = tid; i < C * K; i += 256) swdw[i / K][i % K] = wdw[i];
    for (int i = tid; i < C; i += 256)     sbdw[i] = bdw[i];

    if (tid < TT * K) {
        const int tt = tid / K, k = tid % K;
        const float o  = off[((size_t)b * T + (t0 + tt)) * K + k];
        float ts = (float)(t0 + tt + k) + o;                 // position in padded axis
        ts = fminf(fmaxf(ts, 0.0f), (float)(TP - 1));
        const float i0f = floorf(ts);
        const int   i0  = (int)i0f;
        int i1 = i0 + 1; if (i1 > TP - 1) i1 = TP - 1;
        sm0[tt][k] = reflect_idx(i0 - PL);
        sm1[tt][k] = reflect_idx(i1 - PL);
        sfr[tt][k] = ts - i0f;
    }
    __syncthreads();

    const float alpha = pa[0];
    float s1 = 0.0f, s2 = 0.0f;

#pragma unroll
    for (int ci = 0; ci < 2; ++ci) {
        const int c = tid + ci * 256;
        for (int tt = 0; tt < TT; ++tt) {
            float acc = sbdw[c];
#pragma unroll
            for (int k = 0; k < K; ++k) {
                const float fr = sfr[tt][k];
                const float x0 = x[((size_t)b * T + sm0[tt][k]) * C + c];
                const float x1 = x[((size_t)b * T + sm1[tt][k]) * C + c];
                acc = fmaf(swdw[c][k], fmaf(fr, x1 - x0, x0), acc);
            }
            const float yv = (acc >= 0.0f) ? acc : alpha * acc;
            y[((size_t)b * T + (t0 + tt)) * C + c] = yv;
            s1 += yv;
            s2 = fmaf(yv, yv, s2);
        }
    }

    sred[tid] = s1; __syncthreads();
    for (int s = 128; s > 0; s >>= 1) { if (tid < s) sred[tid] += sred[tid + s]; __syncthreads(); }
    if (tid == 0) atomicAdd(&stats[2 * b + 0], sred[0]);
    __syncthreads();
    sred[tid] = s2; __syncthreads();
    for (int s = 128; s > 0; s >>= 1) { if (tid < s) sred[tid] += sred[tid + s]; __syncthreads(); }
    if (tid == 0) atomicAdd(&stats[2 * b + 1], sred[0]);
}

// ---------------------------------------------------------------------------
// Kernel 3: normalize + gamma/beta, convert to f16 (4 elements/thread)
// ---------------------------------------------------------------------------
__global__ __launch_bounds__(256) void k_norm(const float* __restrict__ y,
                                              const float* __restrict__ stats,
                                              const float* __restrict__ gamma,
                                              const float* __restrict__ beta,
                                              _Float16* __restrict__ yh) {
    const size_t idx = (size_t)blockIdx.x * 256 + threadIdx.x;
    const size_t e   = idx * 4;
    const int b = (int)(e / ((size_t)T * C));
    const int c = (int)(e % C);

    const float inv_n = 1.0f / (float)(T * C);
    const float mu    = stats[2 * b + 0] * inv_n;
    const float var   = stats[2 * b + 1] * inv_n - mu * mu;
    const float rstd  = rsqrtf(var + 1e-8f);

    const float4 v = *reinterpret_cast<const float4*>(y + e);
    union { _Float16 h[4]; uint2 u; } pk;
    pk.h[0] = (_Float16)(gamma[c + 0] * (v.x - mu) * rstd + beta[c + 0]);
    pk.h[1] = (_Float16)(gamma[c + 1] * (v.y - mu) * rstd + beta[c + 1]);
    pk.h[2] = (_Float16)(gamma[c + 2] * (v.z - mu) * rstd + beta[c + 2]);
    pk.h[3] = (_Float16)(gamma[c + 3] * (v.w - mu) * rstd + beta[c + 3]);
    reinterpret_cast<uint2*>(yh)[idx] = pk.u;
}

// ---------------------------------------------------------------------------
// Kernel 4: out = yh (M x C, f16) @ w_pw^T via v_wmma_f32_16x16x32_f16.
// Block tile 128(M) x 128(N); 8 waves as 2(M) x 4(N); wave tile 64x32 = 4x2
// WMMA accumulators.  A/B fragments both use contiguous K order (k = 16*hi + s).
// ---------------------------------------------------------------------------
__global__ __launch_bounds__(256) void k_gemm(const _Float16* __restrict__ yh,
                                              const _Float16* __restrict__ wh,
                                              float* __restrict__ out) {
    const int wave  = threadIdx.x >> 5;
    const int lane  = threadIdx.x & 31;
    const int waveM = wave & 1;
    const int waveN = wave >> 1;
    const int rowBase = blockIdx.x * 128 + waveM * 64;
    const int colBase = blockIdx.y * 128 + waveN * 32;
    const int lm = lane & 15;
    const int hi = lane >> 4;     // lane group (0: lanes 0-15, 1: lanes 16-31)

    v8f acc[4][2];
#pragma unroll
    for (int sm = 0; sm < 4; ++sm)
#pragma unroll
        for (int sn = 0; sn < 2; ++sn) {
            v8f z = {0.f, 0.f, 0.f, 0.f, 0.f, 0.f, 0.f, 0.f};
            acc[sm][sn] = z;
        }

    const _Float16* aBase[4];
#pragma unroll
    for (int sm = 0; sm < 4; ++sm)
        aBase[sm] = yh + (size_t)(rowBase + sm * 16 + lm) * C;
    const _Float16* bBase[2];
#pragma unroll
    for (int sn = 0; sn < 2; ++sn)
        bBase[sn] = wh + (size_t)(colBase + sn * 16 + lm) * C;

    for (int kb = 0; kb < C; kb += 32) {
        if (kb + 64 < C) {                       // gfx1250 global_prefetch path
            __builtin_prefetch(aBase[0] + kb + 64, 0, 1);
            __builtin_prefetch(bBase[0] + kb + 64, 0, 1);
        }
        v16h a[4], bf[2];
#pragma unroll
        for (int sm = 0; sm < 4; ++sm) {
            // A 16x32 f16 layout: lanes 0-15 hold K {0..15}, lanes 16-31 {16..31}
            a[sm] = *reinterpret_cast<const v16h*>(aBase[sm] + kb + hi * 16);
        }
#pragma unroll
        for (int sn = 0; sn < 2; ++sn) {
            // B 32x16 f16 layout: lane = N (+16 for K-section), 16 contiguous K per lane.
            // wh is (OUT, C) row-major -> K(=c) contiguous: single 32B load.
            bf[sn] = *reinterpret_cast<const v16h*>(bBase[sn] + kb + hi * 16);
        }
#pragma unroll
        for (int sm = 0; sm < 4; ++sm)
#pragma unroll
            for (int sn = 0; sn < 2; ++sn)
                acc[sm][sn] = __builtin_amdgcn_wmma_f32_16x16x32_f16(
                    false, a[sm], false, bf[sn], (short)0, acc[sm][sn], false, false);
    }

#pragma unroll
    for (int sm = 0; sm < 4; ++sm)
#pragma unroll
        for (int sn = 0; sn < 2; ++sn) {
            const int col  = colBase + sn * 16 + lm;
            const int row0 = rowBase + sm * 16 + hi * 8;
#pragma unroll
            for (int r = 0; r < 8; ++r)
                out[(size_t)(row0 + r) * OUT + col] = acc[sm][sn][r];
        }
}

// ---------------------------------------------------------------------------
extern "C" void kernel_launch(void* const* d_in, const int* in_sizes, int n_in,
                              void* d_out, int out_size, void* d_ws, size_t ws_size,
                              hipStream_t stream) {
    const float* x     = (const float*)d_in[0];
    const float* wod   = (const float*)d_in[1];
    const float* wop   = (const float*)d_in[2];
    const float* wdw   = (const float*)d_in[3];
    const float* bdw   = (const float*)d_in[4];
    const float* pa    = (const float*)d_in[5];
    const float* gamma = (const float*)d_in[6];
    const float* beta  = (const float*)d_in[7];
    const float* wpw   = (const float*)d_in[8];
    float* out = (float*)d_out;

    // Workspace layout
    char* ws = (char*)d_ws;
    float*    stats = (float*)ws;                                     // 8 floats = 32 B
    float*    off   = (float*)(ws + 256);                             // B*T*K*4 = 720896
    _Float16* wh    = (_Float16*)(ws + 256 + 720896);                 // OUT*C*2 = 524288
    _Float16* yh    = (_Float16*)(ws + 256 + 720896 + 524288);        // B*T*C*2 = 16777216
    float*    yscr  = out;   // y (f32) staged in d_out, overwritten by k_gemm last

    k_prep   <<<(OUT * C + 255) / 256, 256, 0, stream>>>(wpw, wh, stats);
    k_offsets<<<B * (T / 64),          256, 0, stream>>>(x, wod, wop, off);
    k_sample <<<B * (T / 16),          256, 0, stream>>>(x, off, wdw, bdw, pa, yscr, stats);
    k_norm   <<<(B * T * C / 4) / 256, 256, 0, stream>>>(yscr, stats, gamma, beta, yh);
    k_gemm   <<<dim3((B * T) / 128, OUT / 128), 256, 0, stream>>>(yh, wh, out);
}
